// BailingMoEDecoderLayer_80762565034607
// MI455X (gfx1250) — matrix operand order repaired and checked by reading, not compiled, not run on hardware
//
#include <hip/hip_runtime.h>
#include <hip/hip_bf16.h>

// ---------------------------------------------------------------------------
// BailingMoE decoder layer for MI455X (gfx1250, wave32, WMMA).
// T=2048 D=2048 H=16 KV=4 HD=128 E=32 K=4 FM=1024 FS=2048 CAP=512
// Bandwidth-bound (~1GB f32 weights @ 23.3TB/s); all GEMMs use
// v_wmma_f32_16x16x32_bf16 with f32 accumulation. LDS layouts are chosen so
// every WMMA fragment load is a contiguous 16/32-byte vector (ds_load_b128),
// never a column-strided scalar chain.
// ---------------------------------------------------------------------------

typedef __attribute__((ext_vector_type(16))) __bf16 v16bf;
typedef __attribute__((ext_vector_type(8)))  __bf16 v8bf;
typedef __attribute__((ext_vector_type(4)))  __bf16 v4bf;
typedef __attribute__((ext_vector_type(8)))  float  v8f;
typedef __attribute__((ext_vector_type(4)))  float  v4f;

__device__ inline __bf16 f2bf(float f) { return (__bf16)f; }

__device__ inline v16bf join16(v8bf lo, v8bf hi) {
    return __builtin_shufflevector(lo, hi, 0,1,2,3,4,5,6,7,8,9,10,11,12,13,14,15);
}

__device__ inline v8f wmma_bf16(v16bf a, v16bf b, v8f c) {
    // (neg_a, A, neg_b, B, c_mod, C, reuse_a, reuse_b)
    return __builtin_amdgcn_wmma_f32_16x16x32_bf16(false, a, false, b, (short)0, c, false, false);
}

#define TT   2048
#define DD   2048
#define HH   16
#define KVH  4
#define HDIM 128
#define EE   32
#define TOPK 4
#define FMID 1024
#define FSH  2048
#define CAPE 512

// ---------------------------------------------------------------------------
// RMSNorm: y = x * rsqrt(mean(x^2)+eps) * w   (f32 in -> bf16 out)
// ---------------------------------------------------------------------------
__global__ __launch_bounds__(256) void rmsnorm_kernel(
    const float* __restrict__ X, const float* __restrict__ W,
    __bf16* __restrict__ Ybf, int Dd)
{
    int t = blockIdx.x;
    const float* x = X + (long)t * Dd;
    __shared__ float red[8];
    float ss = 0.f;
    for (int d = threadIdx.x; d < Dd; d += 256) { float v = x[d]; ss += v * v; }
    for (int o = 16; o > 0; o >>= 1) ss += __shfl_xor(ss, o, 32);
    if ((threadIdx.x & 31) == 0) red[threadIdx.x >> 5] = ss;
    __syncthreads();
    if (threadIdx.x < 8) {
        float v = red[threadIdx.x];
        for (int o = 4; o > 0; o >>= 1) v += __shfl_xor(v, o, 8);
        red[threadIdx.x] = v;
    }
    __syncthreads();
    float inv = rsqrtf(red[0] / (float)Dd + 1e-6f);
    for (int d = threadIdx.x; d < Dd; d += 256) {
        Ybf[(long)t * Dd + d] = f2bf(x[d] * inv * W[d]);
    }
}

// ---------------------------------------------------------------------------
// Generic batched WMMA GEMM: C[b] = A[b](bf16, MxK) * B[b](f32, KxN) [+ Cadd]
// Block: 256 thr = 8 waves; tile 64(M)x64(N); per-wave 16x32 output.
// BsT stored [n][k] so B-fragment reads are contiguous 32B vectors.
// ---------------------------------------------------------------------------
__global__ __launch_bounds__(256) void wmma_gemm_kernel(
    const __bf16* __restrict__ A, long sA,
    const float*  __restrict__ B, long sB,
    float* __restrict__ C, long sC,
    const float* __restrict__ Cadd,
    const int* __restrict__ counts,
    int M, int N, int Kd)
{
    int batch = blockIdx.z;
    int Mb = M;
    if (counts) { int c = counts[batch]; Mb = (c < M ? c : M); }
    int m0 = blockIdx.y * 64;
    int n0 = blockIdx.x * 64;
    if (m0 >= Mb) return;
    A += (long)batch * sA;
    B += (long)batch * sB;
    C += (long)batch * sC;

    __shared__ alignas(32) __bf16 As[64][32];    // [m][k]
    __shared__ alignas(32) __bf16 BsT[64][32];   // [n][k] (transposed)

    int tid  = threadIdx.x;
    int wave = tid >> 5;
    int lane = tid & 31;
    int wm = wave >> 1;   // 0..3
    int wn = wave & 1;    // 0..1

    int arow = tid >> 2;             // 0..63
    int acol = (tid & 3) * 8;        // 0,8,16,24
    int brow = tid >> 3;             // 0..31  (k)
    int bcol = (tid & 7) * 8;        // 0..56  (n)

    int frow = wm * 16 + (lane & 15);
    int fkh  = (lane >> 4) * 16;
    int fklo = (lane >> 4) * 16;
    int fc0  = wn * 32 + (lane & 15);

    v8f acc0 = {}; v8f acc1 = {};

    for (int kk = 0; kk < Kd; kk += 32) {
        // stage A (64x32 bf16): one 16B vector per thread
        {
            int gr = m0 + arow;
            if (gr < Mb) {
                v8bf av = *(const v8bf*)(A + (long)gr * Kd + kk + acol);
                *(v8bf*)&As[arow][acol] = av;
            } else {
                v8bf z = {};
                *(v8bf*)&As[arow][acol] = z;
            }
        }
        // stage B (32x64 f32, coalesced) -> BsT[n][k] (strided b16 stores)
        {
            const float* bp = B + (long)(kk + brow) * N + n0 + bcol;
            v4f b0 = *(const v4f*)bp;
            v4f b1 = *(const v4f*)(bp + 4);
            #pragma unroll
            for (int i = 0; i < 4; ++i) BsT[bcol + i][brow]     = f2bf(b0[i]);
            #pragma unroll
            for (int i = 0; i < 4; ++i) BsT[bcol + 4 + i][brow] = f2bf(b1[i]);
            if (kk + 32 < Kd)
                __builtin_prefetch(bp + (long)32 * N, 0, 1);  // global_prefetch_b8
        }
        __syncthreads();

        // A fragment: two contiguous 16B loads
        v8bf alo = *(const v8bf*)&As[frow][fkh];
        v8bf ahi = *(const v8bf*)&As[frow][fkh + 8];
        v16bf af = join16(alo, ahi);
        // B fragments: contiguous 32B loads
        v16bf b0 = *(const v16bf*)&BsT[fc0][fklo];
        v16bf b1 = *(const v16bf*)&BsT[fc0 + 16][fklo];

        acc0 = wmma_bf16(af, b0, acc0);
        acc1 = wmma_bf16(af, b1, acc1);
        __syncthreads();
    }

    int nb = n0 + wn * 32 + (lane & 15);
    int mb = m0 + wm * 16 + (lane >> 4) * 8;
    #pragma unroll
    for (int r = 0; r < 8; ++r) {
        int gm = mb + r;
        if (gm < Mb) {
            long i0 = (long)gm * N + nb;
            float v0 = acc0[r], v1 = acc1[r];
            if (Cadd) { v0 += Cadd[i0]; v1 += Cadd[i0 + 16]; }
            C[i0] = v0; C[i0 + 16] = v1;
        }
    }
}

// ---------------------------------------------------------------------------
// Per-(token, head) RMSNorm over HD=128 then RoPE (in place, f32).
// ---------------------------------------------------------------------------
__global__ __launch_bounds__(128) void qknorm_rope_kernel(
    float* __restrict__ X, const float* __restrict__ W,
    const int* __restrict__ positions, int NH)
{
    int t  = blockIdx.x;
    int hh = blockIdx.y;
    float* x = X + ((long)t * NH + hh) * HDIM;
    int tid = threadIdx.x;
    __shared__ float sx[HDIM];
    __shared__ float red[4];
    float v = x[tid];
    float ss = v * v;
    for (int o = 16; o > 0; o >>= 1) ss += __shfl_xor(ss, o, 32);
    if ((tid & 31) == 0) red[tid >> 5] = ss;
    __syncthreads();
    float tot = red[0] + red[1] + red[2] + red[3];
    float inv = rsqrtf(tot / 128.0f + 1e-6f);
    sx[tid] = v * inv * W[tid];
    __syncthreads();
    if (tid < 64) {
        float p = (float)positions[t];
        float freq = __powf(1.0e6f, -((float)tid) / 64.0f);
        float s, c;
        __sincosf(p * freq, &s, &c);
        float x1 = sx[tid], x2 = sx[tid + 64];
        x[tid]      = x1 * c - x2 * s;
        x[tid + 64] = x1 * s + x2 * c;
    }
}

// ---------------------------------------------------------------------------
// Flash attention (causal, GQA 16Q/4KV, HD=128) with WMMA for QK^T and PV.
// Block: 128 thr = 4 waves; 64 query rows per block, one head per blockIdx.y.
// Vs stored transposed [hd][key] so PV B-fragments are contiguous vectors.
// ---------------------------------------------------------------------------
__global__ __launch_bounds__(128) void flash_attn_kernel(
    const float* __restrict__ Q,   // T x H x 128   (rope'd)
    const float* __restrict__ Kg,  // T x KV x 128  (rope'd)
    const float* __restrict__ Vg,  // T x KV x 128
    __bf16* __restrict__ O)        // T x (H*128)
{
    int h   = blockIdx.y;
    int kvh = h / (HH / KVH);
    int q0  = blockIdx.x * 64;
    int tid = threadIdx.x;
    int wave = tid >> 5;
    int lane = tid & 31;

    __shared__ alignas(32) __bf16 Qs[64][HDIM];    // [q][hd]
    __shared__ alignas(32) __bf16 Ks[32][HDIM];    // [key][hd]
    __shared__ alignas(32) __bf16 VsT[HDIM][32];   // [hd][key] (transposed)
    __shared__ alignas(32) __bf16 Ps[4][16][32];   // per-wave P tile

    const float scale = 0.088388347648318447f; // 128^-0.5

    // stage Q scaled: 4 contiguous elems per thread per step
    for (int e4 = tid * 4; e4 < 64 * HDIM; e4 += 128 * 4) {
        int r = e4 >> 7, c = e4 & 127;
        v4f qv = *(const v4f*)&Q[((long)(q0 + r) * HH + h) * HDIM + c];
        v4bf qb;
        #pragma unroll
        for (int i = 0; i < 4; ++i) qb[i] = f2bf(qv[i] * scale);
        *(v4bf*)&Qs[r][c] = qb;
    }
    __syncthreads();

    v8f acc[8] = {};
    float rm[8], rl[8];
    #pragma unroll
    for (int r = 0; r < 8; ++r) { rm[r] = -3.0e38f; rl[r] = 0.f; }

    int lrow = (lane >> 4) * 8;
    int ncol = lane & 15;
    int klo  = (lane >> 4) * 16;

    for (int kb = 0; kb < q0 + 64; kb += 32) {
        // stage K (row-major) and V (transposed)
        for (int e4 = tid * 4; e4 < 32 * HDIM; e4 += 128 * 4) {
            int r = e4 >> 7, c = e4 & 127;
            long gi = ((long)(kb + r) * KVH + kvh) * HDIM + c;
            v4f kv = *(const v4f*)&Kg[gi];
            v4f vv = *(const v4f*)&Vg[gi];
            v4bf kbv;
            #pragma unroll
            for (int i = 0; i < 4; ++i) kbv[i] = f2bf(kv[i]);
            *(v4bf*)&Ks[r][c] = kbv;
            #pragma unroll
            for (int i = 0; i < 4; ++i) VsT[c + i][r] = f2bf(vv[i]);
        }
        __syncthreads();

        // S = Q K^T : two 16x16 tiles (keys kb..+15, kb+16..+31)
        v8f s0 = {}, s1 = {};
        {
            int qrow = wave * 16 + (lane & 15);
            int kh   = (lane >> 4) * 16;
            int key0 = lane & 15;
            #pragma unroll
            for (int kk = 0; kk < HDIM; kk += 32) {
                v8bf alo = *(const v8bf*)&Qs[qrow][kk + kh];
                v8bf ahi = *(const v8bf*)&Qs[qrow][kk + kh + 8];
                v16bf af = join16(alo, ahi);
                v16bf b0 = *(const v16bf*)&Ks[key0][kk + klo];
                v16bf b1 = *(const v16bf*)&Ks[key0 + 16][kk + klo];
                s0 = wmma_bf16(af, b0, s0);
                s1 = wmma_bf16(af, b1, s1);
            }
        }

        // causal mask + online softmax
        #pragma unroll
        for (int r = 0; r < 8; ++r) {
            int qrow = q0 + wave * 16 + lrow + r;
            if (kb + ncol      > qrow) s0[r] = -1e30f;
            if (kb + 16 + ncol > qrow) s1[r] = -1e30f;
        }
        #pragma unroll
        for (int r = 0; r < 8; ++r) {
            float mx = fmaxf(s0[r], s1[r]);
            for (int o = 8; o > 0; o >>= 1) mx = fmaxf(mx, __shfl_xor(mx, o, 32));
            float mnew = fmaxf(rm[r], mx);
            float sc = __expf(rm[r] - mnew);
            rm[r] = mnew;
            float p0 = __expf(s0[r] - mnew);
            float p1 = __expf(s1[r] - mnew);
            float rs = p0 + p1;
            for (int o = 8; o > 0; o >>= 1) rs += __shfl_xor(rs, o, 32);
            rl[r] = rl[r] * sc + rs;
            s0[r] = p0; s1[r] = p1;
            #pragma unroll
            for (int nt = 0; nt < 8; ++nt) acc[nt][r] *= sc;
        }

        // stage P row-major for A-fragment reload
        #pragma unroll
        for (int r = 0; r < 8; ++r) {
            Ps[wave][lrow + r][ncol]      = f2bf(s0[r]);
            Ps[wave][lrow + r][16 + ncol] = f2bf(s1[r]);
        }
        __syncthreads();

        // O += P @ V : A = Ps[wave] (16x32), B from VsT (contiguous)
        v16bf pf;
        {
            int row = lane & 15;
            int kh  = (lane >> 4) * 16;
            v8bf plo = *(const v8bf*)&Ps[wave][row][kh];
            v8bf phi = *(const v8bf*)&Ps[wave][row][kh + 8];
            pf = join16(plo, phi);
        }
        #pragma unroll
        for (int nt = 0; nt < 8; ++nt) {
            int col = nt * 16 + (lane & 15);
            v16bf bv = *(const v16bf*)&VsT[col][klo];
            acc[nt] = wmma_bf16(pf, bv, acc[nt]);
        }
        __syncthreads();
    }

    #pragma unroll
    for (int r = 0; r < 8; ++r) {
        float invl = 1.0f / rl[r];
        int gm = q0 + wave * 16 + lrow + r;
        #pragma unroll
        for (int nt = 0; nt < 8; ++nt)
            O[(long)gm * (HH * HDIM) + h * HDIM + nt * 16 + (lane & 15)] =
                f2bf(acc[nt][r] * invl);
    }
}

// ---------------------------------------------------------------------------
// Router: sigmoid(x @ Wr), top-4, normalized weights, capacity slots.
// ---------------------------------------------------------------------------
__global__ void zero_counts_kernel(int* counts) {
    if (threadIdx.x < EE) counts[threadIdx.x] = 0;
}

__global__ __launch_bounds__(256) void router_topk_kernel(
    const __bf16* __restrict__ X, const float* __restrict__ Wr,
    int* __restrict__ tids, float* __restrict__ tw, int* __restrict__ pos,
    int* __restrict__ counts, int Dd)
{
    int t = blockIdx.x;
    int tid = threadIdx.x;
    int e = tid >> 3;      // 0..31
    int part = tid & 7;
    const __bf16* x = X + (long)t * Dd;
    float sum = 0.f;
    for (int d = part; d < Dd; d += 8) sum += (float)x[d] * Wr[(long)d * EE + e];
    for (int o = 1; o < 8; o <<= 1) sum += __shfl_xor(sum, o, 32);
    __shared__ float sc[EE];
    if (part == 0) sc[e] = 1.0f / (1.0f + __expf(-sum));
    __syncthreads();
    if (tid == 0) {
        float w[TOPK]; int id[TOPK];
        bool used[EE];
        for (int i = 0; i < EE; ++i) used[i] = false;
        float wsum = 0.f;
        for (int k = 0; k < TOPK; ++k) {
            float best = -1.f; int bi = 0;
            for (int i = 0; i < EE; ++i)
                if (!used[i] && sc[i] > best) { best = sc[i]; bi = i; }
            used[bi] = true; w[k] = best; id[k] = bi; wsum += best;
        }
        for (int k = 0; k < TOPK; ++k) {
            int p = atomicAdd(&counts[id[k]], 1);
            tids[t * TOPK + k] = id[k];
            tw[t * TOPK + k]   = w[k] / wsum;
            pos[t * TOPK + k]  = p;
        }
    }
}

__global__ __launch_bounds__(256) void moe_scatter_kernel(
    const __bf16* __restrict__ X, const int* __restrict__ tids,
    const int* __restrict__ pos, __bf16* __restrict__ ebuf, int Dd)
{
    int slot = blockIdx.x;         // t*4+k
    int t = slot >> 2;
    int e = tids[slot];
    int p = pos[slot];
    if (p >= CAPE) return;
    const __bf16* src = X + (long)t * Dd;
    __bf16* dst = ebuf + ((long)e * CAPE + p) * Dd;
    for (int d = threadIdx.x * 8; d < Dd; d += 256 * 8) {
        v8bf v = *(const v8bf*)(src + d);
        *(v8bf*)(dst + d) = v;
    }
}

__global__ void silu_mul_kernel(const float* __restrict__ g,
                                const float* __restrict__ u,
                                __bf16* __restrict__ out, long n)
{
    long i = (long)blockIdx.x * blockDim.x + threadIdx.x;
    if (i < n) {
        float gv = g[i];
        float s = gv / (1.0f + __expf(-gv));
        out[i] = f2bf(s * u[i]);
    }
}

__global__ __launch_bounds__(256) void moe_gather_kernel(
    const float* __restrict__ eo, const int* __restrict__ tids,
    const int* __restrict__ pos, const float* __restrict__ tw,
    float* __restrict__ out, int Dd)
{
    int t = blockIdx.x;
    for (int d = threadIdx.x; d < Dd; d += 256) {
        float acc = out[(long)t * Dd + d];
        #pragma unroll
        for (int k = 0; k < TOPK; ++k) {
            int e = tids[t * TOPK + k];
            int p = pos[t * TOPK + k];
            if (p < CAPE)
                acc += eo[((long)e * CAPE + p) * Dd + d] * tw[t * TOPK + k];
        }
        out[(long)t * Dd + d] = acc;
    }
}

// ---------------------------------------------------------------------------
extern "C" void kernel_launch(void* const* d_in, const int* in_sizes, int n_in,
                              void* d_out, int out_size, void* d_ws, size_t ws_size,
                              hipStream_t stream) {
    const float* hidden   = (const float*)d_in[0];
    const int*   positions= (const int*)  d_in[1];
    const float* ln1_w    = (const float*)d_in[2];
    const float* wq       = (const float*)d_in[3];
    const float* wk       = (const float*)d_in[4];
    const float* wv       = (const float*)d_in[5];
    const float* qn_w     = (const float*)d_in[6];
    const float* kn_w     = (const float*)d_in[7];
    const float* wo       = (const float*)d_in[8];
    const float* ln2_w    = (const float*)d_in[9];
    const float* router_w = (const float*)d_in[10];
    const float* w_gate   = (const float*)d_in[11];
    const float* w_up     = (const float*)d_in[12];
    const float* w_down   = (const float*)d_in[13];
    const float* ws_gate  = (const float*)d_in[14];
    const float* ws_up    = (const float*)d_in[15];
    const float* ws_down  = (const float*)d_in[16];
    float* out = (float*)d_out;

    char* ws = (char*)d_ws;
    size_t off = 0;
    auto alloc = [&](size_t bytes) -> void* {
        void* p = ws + off;
        off += (bytes + 255) & ~(size_t)255;
        return p;
    };

    __bf16* h1    = (__bf16*)alloc((size_t)TT * DD * 2);          // ln1(x) bf16
    float*  qf    = (float*) alloc((size_t)TT * HH * HDIM * 4);   // q
    float*  kf    = (float*) alloc((size_t)TT * KVH * HDIM * 4);  // k
    float*  vf    = (float*) alloc((size_t)TT * KVH * HDIM * 4);  // v
    __bf16* obf   = (__bf16*)alloc((size_t)TT * HH * HDIM * 2);   // attn out
    float*  h2    = (float*) alloc((size_t)TT * DD * 4);          // o@wo + resid
    __bf16* x2    = (__bf16*)alloc((size_t)TT * DD * 2);          // ln2 bf16
    float*  gbuf  = (float*) alloc((size_t)TT * FSH * 4);         // shared gate
    float*  ubuf  = (float*) alloc((size_t)TT * FSH * 4);         // shared up
    __bf16* sact  = (__bf16*)alloc((size_t)TT * FSH * 2);         // silu*up
    int*    tids  = (int*)   alloc((size_t)TT * TOPK * 4);
    float*  twb   = (float*) alloc((size_t)TT * TOPK * 4);
    int*    posb  = (int*)   alloc((size_t)TT * TOPK * 4);
    int*    counts= (int*)   alloc((size_t)EE * 4);
    __bf16* ebuf  = (__bf16*)alloc((size_t)EE * CAPE * DD * 2);
    float*  eg    = (float*) alloc((size_t)EE * CAPE * FMID * 4);
    float*  eu    = (float*) alloc((size_t)EE * CAPE * FMID * 4);
    __bf16* eact  = (__bf16*)alloc((size_t)EE * CAPE * FMID * 2);
    float*  eo    = (float*) alloc((size_t)EE * CAPE * DD * 4);

    zero_counts_kernel<<<1, 32, 0, stream>>>(counts);

    // ln1
    rmsnorm_kernel<<<TT, 256, 0, stream>>>(hidden, ln1_w, h1, DD);

    // QKV projections (WMMA)
    wmma_gemm_kernel<<<dim3(HH * HDIM / 64, TT / 64, 1), 256, 0, stream>>>(
        h1, 0, wq, 0, qf, 0, nullptr, nullptr, TT, HH * HDIM, DD);
    wmma_gemm_kernel<<<dim3(KVH * HDIM / 64, TT / 64, 1), 256, 0, stream>>>(
        h1, 0, wk, 0, kf, 0, nullptr, nullptr, TT, KVH * HDIM, DD);
    wmma_gemm_kernel<<<dim3(KVH * HDIM / 64, TT / 64, 1), 256, 0, stream>>>(
        h1, 0, wv, 0, vf, 0, nullptr, nullptr, TT, KVH * HDIM, DD);

    // QK-norm + RoPE
    qknorm_rope_kernel<<<dim3(TT, HH),  128, 0, stream>>>(qf, qn_w, positions, HH);
    qknorm_rope_kernel<<<dim3(TT, KVH), 128, 0, stream>>>(kf, kn_w, positions, KVH);

    // attention (WMMA flash, causal, GQA)
    flash_attn_kernel<<<dim3(TT / 64, HH), 128, 0, stream>>>(qf, kf, vf, obf);

    // output projection + residual
    wmma_gemm_kernel<<<dim3(DD / 64, TT / 64, 1), 256, 0, stream>>>(
        obf, 0, wo, 0, h2, 0, hidden, nullptr, TT, DD, HH * HDIM);

    // ln2
    rmsnorm_kernel<<<TT, 256, 0, stream>>>(h2, ln2_w, x2, DD);

    // routing + scatter
    router_topk_kernel<<<TT, 256, 0, stream>>>(x2, router_w, tids, twb, posb, counts, DD);
    moe_scatter_kernel<<<TT * TOPK, 256, 0, stream>>>(x2, tids, posb, ebuf, DD);

    // shared expert FFN
    wmma_gemm_kernel<<<dim3(FSH / 64, TT / 64, 1), 256, 0, stream>>>(
        x2, 0, ws_gate, 0, gbuf, 0, nullptr, nullptr, TT, FSH, DD);
    wmma_gemm_kernel<<<dim3(FSH / 64, TT / 64, 1), 256, 0, stream>>>(
        x2, 0, ws_up, 0, ubuf, 0, nullptr, nullptr, TT, FSH, DD);
    {
        long n = (long)TT * FSH;
        silu_mul_kernel<<<(unsigned)((n + 255) / 256), 256, 0, stream>>>(gbuf, ubuf, sact, n);
    }
    // shared down -> d_out (beta=0)
    wmma_gemm_kernel<<<dim3(DD / 64, TT / 64, 1), 256, 0, stream>>>(
        sact, 0, ws_down, 0, out, 0, nullptr, nullptr, TT, DD, FSH);

    // expert FFNs (batched over 32 experts, capacity-limited rows)
    wmma_gemm_kernel<<<dim3(FMID / 64, CAPE / 64, EE), 256, 0, stream>>>(
        ebuf, (long)CAPE * DD, w_gate, (long)DD * FMID, eg, (long)CAPE * FMID,
        nullptr, counts, CAPE, FMID, DD);
    wmma_gemm_kernel<<<dim3(FMID / 64, CAPE / 64, EE), 256, 0, stream>>>(
        ebuf, (long)CAPE * DD, w_up, (long)DD * FMID, eu, (long)CAPE * FMID,
        nullptr, counts, CAPE, FMID, DD);
    {
        long n = (long)EE * CAPE * FMID;
        silu_mul_kernel<<<(unsigned)((n + 255) / 256), 256, 0, stream>>>(eg, eu, eact, n);
    }
    wmma_gemm_kernel<<<dim3(DD / 64, CAPE / 64, EE), 256, 0, stream>>>(
        eact, (long)CAPE * FMID, w_down, (long)FMID * DD, eo, (long)CAPE * DD,
        nullptr, counts, CAPE, DD, FMID);

    // combine: out = shared + sum_k tw * expert_out
    moe_gather_kernel<<<TT, 256, 0, stream>>>(eo, tids, posb, twb, out, DD);

    (void)in_sizes; (void)n_in; (void)out_size; (void)ws_size;
}
